// LoroSparseLinear_78486232367566
// MI455X (gfx1250) — compile-verified, matrix operand order checked by reading, not run on hardware
//
#include <hip/hip_runtime.h>
#include <hip/hip_bf16.h>

typedef __attribute__((ext_vector_type(16))) _Float16 v16h;
typedef __attribute__((ext_vector_type(4)))  _Float16 v4h;
typedef __attribute__((ext_vector_type(2)))  _Float16 v2h;
typedef __attribute__((ext_vector_type(8)))  float    v8f;
typedef _Float16 half_t;

#define IN_F    4096
#define OUT_F   4096
#define RANK    64
#define M_TOTAL 8192   // B*S = 4*2048

// ---------------------------------------------------------------------------
// Weight prep: 2:4 soft threshold per group of 4 along last dim, scale, cast f16.
// t = 2nd-smallest |w| in the group; out = sign(w)*max(|w|-t,0)*scale.
// ---------------------------------------------------------------------------
__global__ void prep_weights_24(const float* __restrict__ w,
                                const float* __restrict__ scale_p,
                                half_t* __restrict__ out, int ngroups) {
    int g = blockIdx.x * blockDim.x + threadIdx.x;
    if (g >= ngroups) return;
    const float4 wv = ((const float4*)w)[g];
    float a = fabsf(wv.x), b = fabsf(wv.y), c = fabsf(wv.z), d = fabsf(wv.w);
    float lo01 = fminf(a, b), hi01 = fmaxf(a, b);
    float lo23 = fminf(c, d), hi23 = fmaxf(c, d);
    // 2nd smallest of 4 = min( max(two small), min(two big) )
    float t = fminf(fmaxf(lo01, lo23), fminf(hi01, hi23));
    float s = *scale_p;
    v4h o;
    o[0] = (half_t)(copysignf(fmaxf(a - t, 0.0f), wv.x) * s);
    o[1] = (half_t)(copysignf(fmaxf(b - t, 0.0f), wv.y) * s);
    o[2] = (half_t)(copysignf(fmaxf(c - t, 0.0f), wv.z) * s);
    o[3] = (half_t)(copysignf(fmaxf(d - t, 0.0f), wv.w) * s);
    ((v4h*)out)[g] = o;
}

__device__ __forceinline__ v16h cvt16_f32_to_f16(float4 f0, float4 f1,
                                                 float4 f2, float4 f3) {
    v16h r;
    r[0]=(half_t)f0.x; r[1]=(half_t)f0.y; r[2]=(half_t)f0.z; r[3]=(half_t)f0.w;
    r[4]=(half_t)f1.x; r[5]=(half_t)f1.y; r[6]=(half_t)f1.z; r[7]=(half_t)f1.w;
    r[8]=(half_t)f2.x; r[9]=(half_t)f2.y; r[10]=(half_t)f2.z; r[11]=(half_t)f2.w;
    r[12]=(half_t)f3.x; r[13]=(half_t)f3.y; r[14]=(half_t)f3.z; r[15]=(half_t)f3.w;
    return r;
}

// ---------------------------------------------------------------------------
// GEMM1: xp[8192x64] = f16(x[8192x4096]) @ f16(sw_in[64x4096])^T
// One block (4 waves) = one 16-row M tile. K=4096 split 4-ways: wave w covers
// k in [w*1024, (w+1)*1024). Partial f32 accumulators reduced through LDS,
// then emitted as f16 xp. 512 blocks -> 2048 streaming waves for the 128 MB
// x read (vs 512 before): enough in-flight loads to cover HBM BDP.
// A lane layout used here (contiguous k, not the ISA table order): lane<16 -> row=lane, K 0..15; lane>=16 -> K 16..31.
// B from row-major (N,K) has the identical per-lane contiguous-16 pattern.
// ---------------------------------------------------------------------------
__global__ void __launch_bounds__(128)
gemm1_wmma(const float* __restrict__ x, const half_t* __restrict__ swin,
           half_t* __restrict__ xp) {
    __shared__ float red[4][16][64];               // 16 KB partial sums

    const int lane  = threadIdx.x & 31;
    const int wave  = threadIdx.x >> 5;            // K-quarter selector
    const int m0    = blockIdx.x * 16;             // 512 tiles of 16 rows
    const int rsel  = lane & 15;                   // row (A) / col-of-N-subtile (B)
    const int koff  = (lane >> 4) << 4;            // 0 or 16
    const int kbeg  = wave * (IN_F / 4);           // 1024-wide K quarter

    const float* xrow = x + (size_t)(m0 + rsel) * IN_F;

    v8f acc[4] = {v8f{}, v8f{}, v8f{}, v8f{}};

    for (int kb = 0; kb < (IN_F / 4) / 32; ++kb) { // 32 K-steps of 32
        const int k = kbeg + kb * 32 + koff;
        const float4* xf = (const float4*)(xrow + k);
        __builtin_prefetch(xrow + k + 512, 0, 0);  // stream-ahead: global_prefetch_b8
        v16h aV = cvt16_f32_to_f16(xf[0], xf[1], xf[2], xf[3]);
#pragma unroll
        for (int nt = 0; nt < 4; ++nt) {
            const v16h bV = *(const v16h*)(swin + (size_t)(nt * 16 + rsel) * IN_F + k);
            acc[nt] = __builtin_amdgcn_wmma_f32_16x16x32_f16(
                false, aV, false, bV, (short)0, acc[nt], false, false);
        }
    }

    // Spill partial tile to LDS.
    // D layout: VGPR v, lanes 0-15 -> row v; lanes 16-31 -> row v+8; col = lane&15
    const int rbase = (lane >> 4) * 8;
    const int colD  = lane & 15;
#pragma unroll
    for (int nt = 0; nt < 4; ++nt)
#pragma unroll
        for (int v = 0; v < 8; ++v)
            red[wave][rbase + v][nt * 16 + colD] = acc[nt][v];

    __syncthreads();

    // Cross-wave reduce: 1024 elements, 128 threads x 2-element packed stores.
#pragma unroll
    for (int i = 0; i < 4; ++i) {
        const int e = (threadIdx.x + i * 128) * 2;     // 0..1022, even
        const int r = e >> 6;
        const int c = e & 63;
        float s0 = red[0][r][c]     + red[1][r][c]     + red[2][r][c]     + red[3][r][c];
        float s1 = red[0][r][c + 1] + red[1][r][c + 1] + red[2][r][c + 1] + red[3][r][c + 1];
        v2h p; p[0] = (half_t)s0; p[1] = (half_t)s1;
        *(v2h*)(xp + (size_t)(m0 + r) * RANK + c) = p;
    }
}

// ---------------------------------------------------------------------------
// GEMM2: out[8192x4096] = (f16(xp[8192x64]) @ f16(sw_out[4096x64])^T + bias)/64
// One wave = 16-row x 64-col tile; K=64 -> 2 WMMA steps per N sub-tile.
// 32768 waves: the 128 MB f32 store side saturates easily.
// ---------------------------------------------------------------------------
__global__ void __launch_bounds__(256)
gemm2_wmma(const half_t* __restrict__ xp, const half_t* __restrict__ swout,
           const float* __restrict__ bias, float* __restrict__ out) {
    const int lane = threadIdx.x & 31;
    const int wave = threadIdx.x >> 5;
    const int tile = blockIdx.x * 8 + wave;        // 32768 tiles (512 M x 64 N-strips)
    const int nstrips = OUT_F / 64;                // 64
    const int mt = tile / nstrips;
    const int ns = tile % nstrips;
    const int m0 = mt * 16;
    const int n0 = ns * 64;
    const int rsel = lane & 15;
    const int koff = (lane >> 4) << 4;

    const half_t* arow = xp + (size_t)(m0 + rsel) * RANK;
    const v16h a0 = *(const v16h*)(arow + koff);        // K 0..31
    const v16h a1 = *(const v16h*)(arow + 32 + koff);   // K 32..63

    v8f acc[4] = {v8f{}, v8f{}, v8f{}, v8f{}};
#pragma unroll
    for (int nt = 0; nt < 4; ++nt) {
        const half_t* bp = swout + (size_t)(n0 + nt * 16 + rsel) * RANK;
        const v16h b0 = *(const v16h*)(bp + koff);
        const v16h b1 = *(const v16h*)(bp + 32 + koff);
        acc[nt] = __builtin_amdgcn_wmma_f32_16x16x32_f16(
            false, a0, false, b0, (short)0, acc[nt], false, false);
        acc[nt] = __builtin_amdgcn_wmma_f32_16x16x32_f16(
            false, a1, false, b1, (short)0, acc[nt], false, false);
    }

    const int rbase = (lane >> 4) * 8;
    const int colD  = lane & 15;
    const float scaling = 1.0f / (float)RANK;
#pragma unroll
    for (int nt = 0; nt < 4; ++nt) {
        const float bv = bias[n0 + nt * 16 + colD];
#pragma unroll
        for (int v = 0; v < 8; ++v)
            out[(size_t)(m0 + rbase + v) * OUT_F + n0 + nt * 16 + colD] =
                (acc[nt][v] + bv) * scaling;
    }
}

// ---------------------------------------------------------------------------
// Launch: prep both weight panels, GEMM1 into ws, GEMM2 into d_out.
// ws layout (halves): [sw_in 64*4096][sw_out 4096*64][xp 8192*64] = 2 MB.
// ---------------------------------------------------------------------------
extern "C" void kernel_launch(void* const* d_in, const int* in_sizes, int n_in,
                              void* d_out, int out_size, void* d_ws, size_t ws_size,
                              hipStream_t stream) {
    const float* x         = (const float*)d_in[0];
    const float* weight_in = (const float*)d_in[1];
    const float* weight_out= (const float*)d_in[2];
    const float* bias      = (const float*)d_in[3];
    const float* scale_in  = (const float*)d_in[4];
    const float* scale_out = (const float*)d_in[5];
    float* out = (float*)d_out;

    half_t* ws     = (half_t*)d_ws;
    half_t* sw_in  = ws;                            // 64*4096
    half_t* sw_out = sw_in  + (size_t)RANK * IN_F;  // 4096*64
    half_t* xp     = sw_out + (size_t)OUT_F * RANK; // 8192*64

    const int g_in  = (RANK * IN_F) / 4;   // 65536 groups
    const int g_out = (OUT_F * RANK) / 4;  // 65536 groups
    prep_weights_24<<<(g_in  + 255) / 256, 256, 0, stream>>>(weight_in,  scale_in,  sw_in,  g_in);
    prep_weights_24<<<(g_out + 255) / 256, 256, 0, stream>>>(weight_out, scale_out, sw_out, g_out);

    // 512 M-tiles, one block each (4 K-split waves) -> 2048 streaming waves
    gemm1_wmma<<<512, 128, 0, stream>>>(x, sw_in, xp);
    // 32768 tiles, 8 waves/block -> 4096 blocks
    gemm2_wmma<<<4096, 256, 0, stream>>>(xp, sw_out, bias, out);
}
